// SuperPoint_4406636446260
// MI455X (gfx1250) — matrix operand order repaired and checked by reading, not compiled, not run on hardware
//
#include <hip/hip_runtime.h>

typedef __attribute__((ext_vector_type(16))) _Float16 v16h;
typedef __attribute__((ext_vector_type(8)))  float    v8f;
typedef __attribute__((ext_vector_type(4)))  unsigned u32x4;
typedef __attribute__((ext_vector_type(8)))  unsigned u32x8;

#define CAP 65536
#define NK  2048

// ---------------------------------------------------------------------------
// conv1a: f32 image (Cin=1) -> f16 NHWC 64ch, 3x3 pad1, ReLU.  Tiny FLOP count.
// ---------------------------------------------------------------------------
__global__ __launch_bounds__(256)
void conv1a_kernel(const float* __restrict__ img, const float* __restrict__ Wt,
                   const float* __restrict__ bias, _Float16* __restrict__ out,
                   int B_, int H, int W_)
{
    __shared__ float wsh[64 * 9];
    int t = threadIdx.x;
    for (int i = t; i < 64 * 9; i += 256) wsh[i] = Wt[i];
    __syncthreads();
    long long gid = (long long)blockIdx.x * 256 + t;
    long long total = (long long)B_ * H * W_;
    if (gid >= total) return;
    int b = (int)(gid / ((long long)H * W_));
    int rem = (int)(gid % ((long long)H * W_));
    int y = rem / W_, x = rem % W_;
    float pix[9];
#pragma unroll
    for (int k = 0; k < 9; ++k) {
        int yy = y + k / 3 - 1, xx = x + k % 3 - 1;
        pix[k] = (yy >= 0 && yy < H && xx >= 0 && xx < W_)
                 ? img[((size_t)b * H + yy) * W_ + xx] : 0.0f;
    }
    _Float16* o = out + (size_t)gid * 64;
    for (int co = 0; co < 64; ++co) {
        float a = bias[co];
#pragma unroll
        for (int k = 0; k < 9; ++k) a += wsh[co * 9 + k] * pix[k];
        o[co] = (_Float16)fmaxf(a, 0.0f);
    }
}

// ---------------------------------------------------------------------------
// Weight repack: f32 OIHW -> f16 in WMMA B-fragment-tile order:
//   dst[((kc*nCT + ct)*32 + lane)*16 + e],  k_local = e + 16*(lane>>4)
// Each lane's 16-f16 fragment is contiguous (32B) -> global_load_b128 x2.
// ---------------------------------------------------------------------------
__global__ void repack_w(const float* __restrict__ Wsrc, _Float16* __restrict__ dst,
                         int Cout, int Cin, int KK, int CoutPad)
{
    int i = blockIdx.x * 256 + threadIdx.x;
    int nCT = CoutPad >> 4;
    int total = KK * Cin * CoutPad;
    if (i >= total) return;
    int e    = i & 15;
    int lane = (i >> 4) & 31;
    int rest = i >> 9;
    int ct   = rest % nCT;
    int kc   = rest / nCT;
    int row  = kc * 32 + ((lane >> 4) << 4) + e;   // k_local = e + 16*half
    int tap  = row / Cin;                          // Cin % 32 == 0
    int ci   = row % Cin;
    int co   = ct * 16 + (lane & 15);
    _Float16 v = (_Float16)0.0f;
    if (co < Cout) v = (_Float16)Wsrc[((size_t)co * Cin + ci) * KK + tap];
    dst[i] = v;
}

// ---------------------------------------------------------------------------
// Generic WMMA conv (3x3 pad1 or 1x1).  NHWC fp16 in/out, fp32 accumulate.
// Macro tile: 128 pixels (8 wide x 16 tall) x 64 out channels; 8 waves,
// each owning a 2x2 grid of 16x16 WMMA tiles (4 v_wmma per K-chunk).
//
// A staging: linear [pixel 0..127][k 0..31] f16 LDS image.  For interior
// tiles a single TENSOR_LOAD_TO_LDS (3-D D#: X=32 ci, Y=8 px, Z=16 py,
// strides Cin / Cin*W, data_size=2B) issued by wave 0 DMAs the whole 8KB
// tile; completion via s_wait_tensorcnt + barrier.  Border tiles use
// per-thread guarded b128 staging.  Per-lane A fragments are then
// 2x ds_load_b128 (row-major K groups of 8 are contiguous per lane).
// ---------------------------------------------------------------------------
__global__ __launch_bounds__(256)
void conv_wmma(const _Float16* __restrict__ in, const _Float16* __restrict__ wB,
               const float* __restrict__ bias, _Float16* __restrict__ out,
               int B_, int H, int W_, int Cin, int Cout, int CoutPad,
               int CStore, int k3, int relu)
{
    __shared__ __align__(16) _Float16 lsA[128 * 32];   // linear [pixel][k], 8KB

    const int t    = threadIdx.x;
    const int lane = t & 31;
    const int wid  = t >> 5;
    const int mt0  = (wid & 3) * 2;    // first of 2 M tiles for this wave
    const int nh   = wid >> 2;         // 32-channel half

    const int nCB = CoutPad >> 6;
    const int nCT = CoutPad >> 4;
    const int b   = blockIdx.z / nCB;
    const int cb  = blockIdx.z % nCB;
    const int x0  = blockIdx.x * 8;
    const int y0  = blockIdx.y * 16;

    v8f acc00 = {}, acc01 = {}, acc10 = {}, acc11 = {};

    const int nk = k3 ? 9 : 1;
    const int nc = Cin >> 5;

    // fallback per-thread staging: 2 x 16B stores into the linear image
    int pS[2], ciS[2], ofS[2];
#pragma unroll
    for (int s = 0; s < 2; ++s) {
        int idx   = t + s * 256;
        int p     = idx >> 2;           // pixel 0..127
        int ciOff = (idx & 3) << 3;     // 0,8,16,24
        pS[s] = p; ciS[s] = ciOff;
        ofS[s] = p * 32 + ciOff;
    }

    // ---- invariant TDM descriptor pieces (D# groups 1..3, ISA 8.4-8.6) ----
    const unsigned uCin = (unsigned)Cin, uW = (unsigned)W_;
    const unsigned s1   = uCin * uW;               // dim1 stride (elements)
    u32x8 g1;
    g1[0] = 0x00010000u;                           // wg_mask=0, data_size=2B
    g1[1] = (uCin & 0xffffu) << 16;                // tensor_dim0[15:0]
    g1[2] = (uCin >> 16) | ((uW & 0xffffu) << 16); // tensor_dim0[31:16] | tensor_dim1[15:0]
    g1[3] = (uW >> 16) | (32u << 16);              // tensor_dim1[31:16] | tile_dim0=32
    g1[4] = 8u | (16u << 16);                      // tile_dim1=8 | tile_dim2=16
    g1[5] = uCin;                                  // tensor_dim0_stride[31:0]
    g1[6] = (s1 & 0xffffu) << 16;                  // stride0[47:32]=0 | stride1[15:0]
    g1[7] = s1 >> 16;                              // tensor_dim1_stride[47:16]
    u32x4 g2;
    g2[0] = (unsigned)H;                           // tensor_dim2
    g2[1] = 0u; g2[2] = 0u; g2[3] = 0u;
    u32x4 g3; g3[0] = 0u; g3[1] = 0u; g3[2] = 0u; g3[3] = 0u;
    const unsigned lds_lo = (unsigned)(__SIZE_TYPE__)&lsA[0]; // LDS byte address

    const int m    = lane & 15;
    const int half = lane >> 4;
    const int ct0  = cb * 4 + nh * 2;

    int gk = 0;
    for (int tap = 0; tap < nk; ++tap) {
        const int dy = k3 ? (tap / 3 - 1) : 0;
        const int dx = k3 ? (tap % 3 - 1) : 0;
        const bool interior = (y0 + dy >= 0) && (y0 + dy + 16 <= H) &&
                              (x0 + dx >= 0) && (x0 + dx + 8 <= W_);
        for (int c = 0; c < nc; ++c, ++gk) {
            if (interior) {
                // ---- async tensor DMA of the whole 16x8x32 tile ----
                if (wid == 0) {
                    unsigned long long ga = (unsigned long long)(__SIZE_TYPE__)
                        (in + (((size_t)b * H + (y0 + dy)) * W_ + (x0 + dx)) * Cin + c * 32);
                    u32x4 g0;
                    g0[0] = 1u;                                  // count=1, user mode
                    g0[1] = lds_lo;                              // lds_addr
                    g0[2] = (unsigned)ga;                        // global_addr[31:0]
                    g0[3] = (unsigned)(ga >> 32) | (2u << 30);   // global_addr[56:32] | type=2
                    asm volatile("tensor_load_to_lds %0, %1, %2, %3"
                                 :: "s"(g0), "s"(g1), "s"(g2), "s"(g3)
                                 : "memory");
                    __builtin_amdgcn_s_wait_tensorcnt(0);
                }
            } else {
                // ---- border: guarded per-thread staging ----
#pragma unroll
                for (int s = 0; s < 2; ++s) {
                    int p  = pS[s];
                    int yy = y0 + (p >> 3) + dy;
                    int xx = x0 + (p & 7) + dx;
                    bool inb = (yy >= 0) && (yy < H) && (xx >= 0) && (xx < W_);
                    uint4 av = {0u, 0u, 0u, 0u};
                    if (inb)
                        av = *(const uint4*)(in + (((size_t)b * H + yy) * W_ + xx) * Cin
                                             + c * 32 + ciS[s]);
                    *(uint4*)&lsA[ofS[s]] = av;
                }
            }
            __syncthreads();

            // ---- B fragments: 2x global b128 from fragment-ordered weights ----
            v16h bf0, bf1;
            {
                const _Float16* pb0 = wB + (((size_t)(gk * nCT + ct0)     * 32 + lane) << 4);
                const _Float16* pb1 = wB + (((size_t)(gk * nCT + ct0 + 1) * 32 + lane) << 4);
                ((uint4*)&bf0)[0] = ((const uint4*)pb0)[0];
                ((uint4*)&bf0)[1] = ((const uint4*)pb0)[1];
                ((uint4*)&bf1)[0] = ((const uint4*)pb1)[0];
                ((uint4*)&bf1)[1] = ((const uint4*)pb1)[1];
            }
            // ---- A fragments: 2x ds b128 from the linear image ----
            // lane (m, half): K range {half*16 .. +15}
            v16h af0, af1;
            {
                const _Float16* pa0 = &lsA[(mt0 * 16 + m) * 32 + half * 16];
                const _Float16* pa1 = pa0 + 16 * 32;
                ((uint4*)&af0)[0] = *(const uint4*)pa0;
                ((uint4*)&af0)[1] = *(const uint4*)(pa0 + 8);
                ((uint4*)&af1)[0] = *(const uint4*)pa1;
                ((uint4*)&af1)[1] = *(const uint4*)(pa1 + 8);
            }
            acc00 = __builtin_amdgcn_wmma_f32_16x16x32_f16(false, af0, false, bf0,
                                                           (short)0, acc00, false, false);
            acc01 = __builtin_amdgcn_wmma_f32_16x16x32_f16(false, af0, false, bf1,
                                                           (short)0, acc01, false, false);
            acc10 = __builtin_amdgcn_wmma_f32_16x16x32_f16(false, af1, false, bf0,
                                                           (short)0, acc10, false, false);
            acc11 = __builtin_amdgcn_wmma_f32_16x16x32_f16(false, af1, false, bf1,
                                                           (short)0, acc11, false, false);
            __syncthreads();
        }
    }

    // ---- store D per 16x16 f32 C/D layout ----
    const int col  = cb * 64 + nh * 32 + m;
    const float b0 = (col      < Cout) ? bias[col]      : 0.0f;
    const float b1 = (col + 16 < Cout) ? bias[col + 16] : 0.0f;
#pragma unroll
    for (int mi = 0; mi < 2; ++mi) {
        v8f a0 = mi ? acc10 : acc00;
        v8f a1 = mi ? acc11 : acc01;
#pragma unroll
        for (int r = 0; r < 8; ++r) {
            int p  = (mt0 + mi) * 16 + r + 8 * half;
            int py = p >> 3, px = p & 7;
            float v0 = a0[r] + b0;
            float v1 = a1[r] + b1;
            if (relu) { v0 = fmaxf(v0, 0.0f); v1 = fmaxf(v1, 0.0f); }
            size_t base = (((size_t)b * H + (y0 + py)) * W_ + (x0 + px)) * (size_t)CStore;
            if (col      < Cout) out[base + col]      = (_Float16)v0;
            if (col + 16 < Cout) out[base + col + 16] = (_Float16)v1;
        }
    }
}

// ---------------------------------------------------------------------------
// 2x2 max pool, NHWC f16
// ---------------------------------------------------------------------------
__global__ void pool2_kernel(const _Float16* __restrict__ in, _Float16* __restrict__ out,
                             int B_, int Ho, int Wo, int C)
{
    size_t i = (size_t)blockIdx.x * 256 + threadIdx.x;
    size_t total = (size_t)B_ * Ho * Wo * C;
    if (i >= total) return;
    int c = (int)(i % C); size_t r = i / C;
    int x = (int)(r % Wo); r /= Wo;
    int y = (int)(r % Ho); int b = (int)(r / Ho);
    int Wi = Wo * 2;
    const _Float16* p = in + (((size_t)b * (Ho * 2) + 2 * y) * Wi + 2 * x) * C + c;
    float v0 = (float)p[0], v1 = (float)p[C];
    float v2 = (float)p[(size_t)Wi * C], v3 = (float)p[(size_t)Wi * C + C];
    out[i] = (_Float16)fmaxf(fmaxf(v0, v1), fmaxf(v2, v3));
}

// ---------------------------------------------------------------------------
// softmax over 65 logits, drop dustbin, pixel-shuffle 8x8 -> full-res scores
// ---------------------------------------------------------------------------
__global__ void softmax_shuffle(const _Float16* __restrict__ sc, float* __restrict__ S, int B_)
{
    int gid = blockIdx.x * 256 + threadIdx.x;
    int total = B_ * 128 * 128;
    if (gid >= total) return;
    int b   = gid / (128 * 128);
    int rem = gid % (128 * 128);
    int hc = rem / 128, wc = rem % 128;
    const _Float16* p = sc + (size_t)gid * 65;
    float v[65];
    float mx = -1e30f;
#pragma unroll
    for (int c = 0; c < 65; ++c) { v[c] = (float)p[c]; mx = fmaxf(mx, v[c]); }
    float sum = 0.0f;
#pragma unroll
    for (int c = 0; c < 65; ++c) { v[c] = __expf(v[c] - mx); sum += v[c]; }
    float inv = 1.0f / sum;
#pragma unroll
    for (int c = 0; c < 64; ++c) {
        int dy = c >> 3, dx = c & 7;
        S[((size_t)b * 1024 + hc * 8 + dy) * 1024 + (wc * 8 + dx)] = v[c] * inv;
    }
}

// ---------------------------------------------------------------------------
// separable 9-wide max pools ("same", clamped) and NMS pointwise ops
// ---------------------------------------------------------------------------
__global__ void mp_h(const float* __restrict__ in, float* __restrict__ out,
                     int B_, int H, int W_)
{
    size_t i = (size_t)blockIdx.x * 256 + threadIdx.x;
    size_t total = (size_t)B_ * H * W_;
    if (i >= total) return;
    int x = (int)(i % W_);
    size_t row = i - x;
    int lo = x - 4 > 0 ? x - 4 : 0;
    int hi = x + 4 < W_ - 1 ? x + 4 : W_ - 1;
    float mv = -1e30f;
    for (int xx = lo; xx <= hi; ++xx) mv = fmaxf(mv, in[row + xx]);
    out[i] = mv;
}

__global__ void mp_v(const float* __restrict__ in, float* __restrict__ out,
                     int B_, int H, int W_)
{
    size_t i = (size_t)blockIdx.x * 256 + threadIdx.x;
    size_t total = (size_t)B_ * H * W_;
    if (i >= total) return;
    int x = (int)(i % W_); size_t r = i / W_;
    int y = (int)(r % H);  int b = (int)(r / H);
    int lo = y - 4 > 0 ? y - 4 : 0;
    int hi = y + 4 < H - 1 ? y + 4 : H - 1;
    size_t base = (size_t)b * H * W_ + x;
    float mv = -1e30f;
    for (int yy = lo; yy <= hi; ++yy) mv = fmaxf(mv, in[base + (size_t)yy * W_]);
    out[i] = mv;
}

__global__ void k_eq_mask(const float* __restrict__ s, const float* __restrict__ mp,
                          float* __restrict__ mask, size_t n)
{
    size_t i = (size_t)blockIdx.x * 256 + threadIdx.x;
    if (i >= n) return;
    mask[i] = (s[i] == mp[i]) ? 1.0f : 0.0f;
}

__global__ void k_supp_scores(const float* __restrict__ suppmp, const float* __restrict__ s,
                              float* __restrict__ ss, size_t n)
{
    size_t i = (size_t)blockIdx.x * 256 + threadIdx.x;
    if (i >= n) return;
    ss[i] = (suppmp[i] > 0.0f) ? 0.0f : s[i];
}

__global__ void k_update_mask(float* __restrict__ mask, const float* __restrict__ suppmp,
                              const float* __restrict__ ss, const float* __restrict__ ssmp,
                              size_t n)
{
    size_t i = (size_t)blockIdx.x * 256 + threadIdx.x;
    if (i >= n) return;
    bool new_max = (ss[i] == ssmp[i]);
    bool supp    = (suppmp[i] > 0.0f);
    if (new_max && !supp) mask[i] = 1.0f;
}

// ---------------------------------------------------------------------------
// top-k: sortable 64-bit keys (value desc, index asc), bitonic sort
// ---------------------------------------------------------------------------
__device__ __forceinline__ unsigned long long make_key(float v, unsigned idx)
{
    unsigned s = __float_as_uint(v);
    unsigned u = (s & 0x80000000u) ? ~s : (s | 0x80000000u);
    return ((unsigned long long)(~u) << 32) | idx;
}

__global__ void init_keys(unsigned long long* __restrict__ keys, int* __restrict__ counts)
{
    int b = blockIdx.y;
    int i = blockIdx.x * 256 + threadIdx.x;
    if (i < CAP) keys[(size_t)b * CAP + i] = make_key(-1.0f, (unsigned)i);
    if (i == 0) counts[b] = 0;
}

__global__ void k_finalize(const float* __restrict__ mask, const float* __restrict__ s,
                           unsigned long long* __restrict__ keys, int* __restrict__ counts,
                           int B_, int H, int W_)
{
    size_t i = (size_t)blockIdx.x * 256 + threadIdx.x;
    size_t total = (size_t)B_ * H * W_;
    if (i >= total) return;
    int x = (int)(i % W_); size_t r = i / W_;
    int y = (int)(r % H);  int b = (int)(r / H);
    float v = (mask[i] != 0.0f) ? s[i] : 0.0f;
    bool border = (y < 4) || (y >= H - 4) || (x < 4) || (x >= W_ - 4);
    if (!border && v > 0.0005f) {
        int slot = atomicAdd(&counts[b], 1);
        if (slot < CAP) keys[(size_t)b * CAP + slot] = make_key(v, (unsigned)(y * W_ + x));
    }
}

__global__ void bitonic_step(unsigned long long* __restrict__ keys, int k, int j)
{
    int b = blockIdx.y;
    unsigned long long* a = keys + (size_t)b * CAP;
    int i = blockIdx.x * 256 + threadIdx.x;
    int ixj = i ^ j;
    if (ixj > i) {
        unsigned long long x = a[i], y = a[ixj];
        bool asc = ((i & k) == 0);
        if ((asc && x > y) || (!asc && x < y)) { a[i] = y; a[ixj] = x; }
    }
}

__global__ void write_topk(const unsigned long long* __restrict__ keys,
                           float* __restrict__ kp, float* __restrict__ kv, int W_)
{
    int b = blockIdx.y;
    int r = blockIdx.x * 256 + threadIdx.x;
    if (r >= NK) return;
    unsigned long long key = keys[(size_t)b * CAP + r];
    unsigned idx = (unsigned)(key & 0xffffffffu);
    unsigned u = ~(unsigned)(key >> 32);
    unsigned sb = (u & 0x80000000u) ? (u & 0x7fffffffu) : ~u;
    float v = __uint_as_float(sb);
    kp[((size_t)b * NK + r) * 2 + 0] = (float)(idx % (unsigned)W_);
    kp[((size_t)b * NK + r) * 2 + 1] = (float)(idx / (unsigned)W_);
    kv[(size_t)b * NK + r] = v;
}

// ---------------------------------------------------------------------------
// per-pixel L2 normalization of descriptor map (f16 NHWC -> f32 NHWC)
// ---------------------------------------------------------------------------
__global__ __launch_bounds__(256)
void l2norm_map(const _Float16* __restrict__ in, float* __restrict__ out)
{
    __shared__ float red[256];
    size_t pix = blockIdx.x;
    int c = threadIdx.x;
    float v = (float)in[pix * 256 + c];
    red[c] = v * v;
    __syncthreads();
    for (int s = 128; s > 0; s >>= 1) {
        if (c < s) red[c] += red[c + s];
        __syncthreads();
    }
    float inv = 1.0f / fmaxf(sqrtf(red[0]), 1e-12f);
    out[pix * 256 + c] = v * inv;
}

// ---------------------------------------------------------------------------
// bilinear descriptor sampling at keypoints + L2 norm -> [b, k, 256]
// ---------------------------------------------------------------------------
__global__ __launch_bounds__(256)
void sample_desc(const float* __restrict__ dmap, const float* __restrict__ kp,
                 float* __restrict__ out)
{
    __shared__ float red[256];
    int b = blockIdx.y;
    int q = blockIdx.x;
    int c = threadIdx.x;
    float x = kp[((size_t)b * NK + q) * 2 + 0];
    float y = kp[((size_t)b * NK + q) * 2 + 1];
    const float scale = 127.0f / 1019.5f;
    float ix = (x - 3.5f) * scale;
    float iy = (y - 3.5f) * scale;
    float x0 = floorf(ix), y0 = floorf(iy);
    float wx = ix - x0, wy = iy - y0;
    int x0i = min(max((int)x0, 0), 127);
    int x1i = min(max((int)x0 + 1, 0), 127);
    int y0i = min(max((int)y0, 0), 127);
    int y1i = min(max((int)y0 + 1, 0), 127);
    const float* base = dmap + (size_t)b * 128 * 128 * 256;
    float d00 = base[((size_t)y0i * 128 + x0i) * 256 + c];
    float d01 = base[((size_t)y0i * 128 + x1i) * 256 + c];
    float d10 = base[((size_t)y1i * 128 + x0i) * 256 + c];
    float d11 = base[((size_t)y1i * 128 + x1i) * 256 + c];
    float v = d00 * (1.0f - wx) * (1.0f - wy) + d01 * wx * (1.0f - wy)
            + d10 * (1.0f - wx) * wy          + d11 * wx * wy;
    red[c] = v * v;
    __syncthreads();
    for (int s = 128; s > 0; s >>= 1) {
        if (c < s) red[c] += red[c + s];
        __syncthreads();
    }
    float inv = 1.0f / fmaxf(sqrtf(red[0]), 1e-12f);
    out[((size_t)b * NK + q) * 256 + c] = v * inv;
}

// ---------------------------------------------------------------------------
// host orchestration
// ---------------------------------------------------------------------------
extern "C" void kernel_launch(void* const* d_in, const int* in_sizes, int n_in,
                              void* d_out, int out_size, void* d_ws, size_t ws_size,
                              hipStream_t stream)
{
    (void)in_sizes; (void)n_in; (void)out_size; (void)ws_size;
    const int B_ = 2;
    const float* img = (const float*)d_in[0];
    const float* Wp[12]; const float* bp[12];
    for (int i = 0; i < 12; ++i) { Wp[i] = (const float*)d_in[1 + 2 * i];
                                   bp[i] = (const float*)d_in[2 + 2 * i]; }

    size_t off = 0;
    char* ws = (char*)d_ws;
    auto alloc = [&](size_t bytes) -> char* {
        char* p = ws + off;
        off += (bytes + 255) & ~(size_t)255;
        return p;
    };
    _Float16* A0        = (_Float16*)alloc((size_t)B_ * 1024 * 1024 * 64 * 2);
    _Float16* A1        = (_Float16*)alloc((size_t)B_ * 1024 * 1024 * 64 * 2);
    _Float16* wbuf      = (_Float16*)alloc((size_t)9 * 128 * 256 * 2);
    _Float16* scoresRaw = (_Float16*)alloc((size_t)B_ * 128 * 128 * 65 * 2);
    _Float16* descH     = (_Float16*)alloc((size_t)B_ * 128 * 128 * 256 * 2);
    float*    dmap      = (float*)   alloc((size_t)B_ * 128 * 128 * 256 * 4);
    float*    S         = (float*)   alloc((size_t)B_ * 1024 * 1024 * 4);
    float*    T         = (float*)   alloc((size_t)B_ * 1024 * 1024 * 4);
    float*    M         = (float*)   alloc((size_t)B_ * 1024 * 1024 * 4);
    float*    M2        = (float*)   alloc((size_t)B_ * 1024 * 1024 * 4);
    float*    MASK      = (float*)   alloc((size_t)B_ * 1024 * 1024 * 4);
    float*    SS        = (float*)   alloc((size_t)B_ * 1024 * 1024 * 4);
    unsigned long long* keys = (unsigned long long*)alloc((size_t)B_ * CAP * 8);
    int*      counts    = (int*)     alloc(256);

    auto run_conv = [&](int li, const _Float16* src, _Float16* dst,
                        int Hc, int Wc, int Cin, int Cout, int CoutPad,
                        int CStore, int k3, int relu) {
        int KK = k3 ? 9 : 1;
        int total = KK * Cin * CoutPad;
        repack_w<<<(total + 255) / 256, 256, 0, stream>>>(Wp[li], wbuf, Cout, Cin, KK, CoutPad);
        dim3 g(Wc / 8, Hc / 16, B_ * (CoutPad / 64));
        conv_wmma<<<g, 256, 0, stream>>>(src, wbuf, bp[li], dst, B_, Hc, Wc,
                                         Cin, Cout, CoutPad, CStore, k3, relu);
    };

    // ---- backbone ----
    conv1a_kernel<<<(B_ * 1024 * 1024 + 255) / 256, 256, 0, stream>>>(
        img, Wp[0], bp[0], A0, B_, 1024, 1024);
    run_conv(1, A0, A1, 1024, 1024, 64, 64, 64, 64, 1, 1);              // conv1b
    pool2_kernel<<<(int)(((size_t)B_ * 512 * 512 * 64 + 255) / 256), 256, 0, stream>>>(
        A1, A0, B_, 512, 512, 64);
    run_conv(2, A0, A1, 512, 512, 64, 64, 64, 64, 1, 1);                // conv2a
    run_conv(3, A1, A0, 512, 512, 64, 64, 64, 64, 1, 1);                // conv2b
    pool2_kernel<<<(int)(((size_t)B_ * 256 * 256 * 64 + 255) / 256), 256, 0, stream>>>(
        A0, A1, B_, 256, 256, 64);
    run_conv(4, A1, A0, 256, 256, 64, 128, 128, 128, 1, 1);             // conv3a
    run_conv(5, A0, A1, 256, 256, 128, 128, 128, 128, 1, 1);            // conv3b
    pool2_kernel<<<(int)(((size_t)B_ * 128 * 128 * 128 + 255) / 256), 256, 0, stream>>>(
        A1, A0, B_, 128, 128, 128);
    run_conv(6, A0, A1, 128, 128, 128, 128, 128, 128, 1, 1);            // conv4a
    run_conv(7, A1, A0, 128, 128, 128, 128, 128, 128, 1, 1);            // conv4b

    // ---- detector head ----
    run_conv(8, A0, A1, 128, 128, 128, 256, 256, 256, 1, 1);            // convPa
    run_conv(9, A1, scoresRaw, 128, 128, 256, 65, 128, 65, 0, 0);       // convPb (1x1)
    softmax_shuffle<<<(B_ * 128 * 128 + 255) / 256, 256, 0, stream>>>(scoresRaw, S, B_);

    // ---- descriptor head ----
    run_conv(10, A0, A1, 128, 128, 128, 256, 256, 256, 1, 1);           // convDa
    run_conv(11, A1, descH, 128, 128, 256, 256, 256, 256, 0, 0);        // convDb (1x1)
    l2norm_map<<<B_ * 128 * 128, 256, 0, stream>>>(descH, dmap);

    // ---- simple_nms (radius 4, two suppression iterations) ----
    const size_t NPIX = (size_t)B_ * 1024 * 1024;
    const int GE = (int)((NPIX + 255) / 256);
    mp_h<<<GE, 256, 0, stream>>>(S, T, B_, 1024, 1024);
    mp_v<<<GE, 256, 0, stream>>>(T, M, B_, 1024, 1024);
    k_eq_mask<<<GE, 256, 0, stream>>>(S, M, MASK, NPIX);
    for (int it = 0; it < 2; ++it) {
        mp_h<<<GE, 256, 0, stream>>>(MASK, T, B_, 1024, 1024);
        mp_v<<<GE, 256, 0, stream>>>(T, M, B_, 1024, 1024);             // M = mp(mask)
        k_supp_scores<<<GE, 256, 0, stream>>>(M, S, SS, NPIX);
        mp_h<<<GE, 256, 0, stream>>>(SS, T, B_, 1024, 1024);
        mp_v<<<GE, 256, 0, stream>>>(T, M2, B_, 1024, 1024);            // M2 = mp(supp_scores)
        k_update_mask<<<GE, 256, 0, stream>>>(MASK, M, SS, M2, NPIX);
    }

    // ---- threshold + border + deterministic top-k ----
    init_keys<<<dim3(CAP / 256, B_), 256, 0, stream>>>(keys, counts);
    k_finalize<<<GE, 256, 0, stream>>>(MASK, S, keys, counts, B_, 1024, 1024);
    for (int k = 2; k <= CAP; k <<= 1)
        for (int j = k >> 1; j > 0; j >>= 1)
            bitonic_step<<<dim3(CAP / 256, B_), 256, 0, stream>>>(keys, k, j);

    float* out_kp = (float*)d_out;                    // [2, 2048, 2]
    float* out_kv = out_kp + (size_t)B_ * NK * 2;     // [2, 2048]
    float* out_de = out_kv + (size_t)B_ * NK;         // [2, 2048, 256]
    write_topk<<<dim3(NK / 256, B_), 256, 0, stream>>>(keys, out_kp, out_kv, 1024);

    // ---- sample descriptors at keypoints ----
    sample_desc<<<dim3(NK, B_), 256, 0, stream>>>(dmap, out_kp, out_de);
}
